// UDUModel_20203526160860
// MI455X (gfx1250) — matrix operand order repaired and checked by reading, not compiled, not run on hardware
//
#include <hip/hip_runtime.h>
#include <math.h>

// ---------------------------------------------------------------------------
// MI455X (gfx1250) implementation of the point-cloud U-Net reference.
// One general strided/batched WMMA GEMM (bf16 in, f32 accum,
// v_wmma_f32_16x16x32_bf16) carries all matmuls: MHA projections, per-head
// QK^T and A*V, conv1d-as-GEMM (shifted accumulating GEMMs), kNN Gram.
// LDS tiles are stored in *fragment order* (two ds_load_b128 per operand),
// and staging is specialized: vectorized b128 global loads + unguarded
// unrolled loops for interior tiles, guarded scalar path only at edges.
// ---------------------------------------------------------------------------

#define BAT 4
#define HEADS 4
#define KNN 10

typedef __attribute__((ext_vector_type(16))) __bf16          v16bf;
typedef __attribute__((ext_vector_type(16))) unsigned short  v16u;
typedef __attribute__((ext_vector_type(8)))  unsigned short  v8u;
typedef __attribute__((ext_vector_type(4)))  unsigned short  v4u;
typedef __attribute__((ext_vector_type(4)))  float           v4f;
typedef __attribute__((ext_vector_type(8)))  float           v8f;

union FragU { v16u u; v16bf b; };

__device__ __forceinline__ unsigned short f2bf(float f) {
  union { float f; unsigned u; } v; v.f = f;
  unsigned r = v.u + 0x7FFFu + ((v.u >> 16) & 1u);   // round-to-nearest-even
  return (unsigned short)(r >> 16);
}

// ---------------------------------------------------------------------------
// General GEMM:  C[z][m,n] (+)= alpha * sum_k A[ofsA(z)+m*rsA+k*csA]
//                                       * B[ofsB(z)+k*rsB+(n+bShift)*csB]
//                ( + bias[n] or bias[m] )
// z = blockIdx.z ; zo=z/inner, zi=z%inner ; ofsX = zo*xOut + zi*xIn
// B column index (n+bShift) outside [0,N) reads as 0  (conv zero-pad).
// Tile: 64x64, 4 waves, K-step 32.
// ---------------------------------------------------------------------------
#define TM 64
#define TN 64
#define TK 32
#define LSTR 40   // LDS row stride in ushorts: 80B, keeps 16B chunk alignment
#define GT 128

__global__ __launch_bounds__(GT) void gemm_wmma(
    const float* __restrict__ A, const float* __restrict__ B,
    const float* __restrict__ bias, float* __restrict__ C,
    int M, int N, int K,
    int rsA, int csA, int rsB, int csB, int ldc,
    long aOut, long aIn, long bOut, long bIn, long cOut, long cIn,
    int inner, float alpha, int bShift, int biasMode, int accum)
{
  // A tile [m][k] (k contiguous); B tile transposed [n][k] (k contiguous)
  __shared__ __attribute__((aligned(16))) unsigned short Al[TM][LSTR];
  __shared__ __attribute__((aligned(16))) unsigned short Bt[TN][LSTR];

  const int z  = blockIdx.z;
  const int zo = z / inner, zi = z % inner;
  const float* Ag = A + zo * aOut + zi * aIn;
  const float* Bg = B + zo * bOut + zi * bIn;
  float*       Cg = C + zo * cOut + zi * cIn;

  const int tid  = threadIdx.x;
  const int lane = tid & 31;
  const int wave = tid >> 5;
  const int half = (lane >> 4) & 1;
  const int l16  = lane & 15;

  const int m0 = blockIdx.y * TM;
  const int n0 = blockIdx.x * TN;

  // block-uniform staging-path selectors
  const bool aFull = (m0 + TM <= M);
  const bool bFull = (n0 + bShift >= 0) && (n0 + TN + bShift <= N);
  const bool aQK = (csA == 1) && ((rsA & 3) == 0) && ((((unsigned long)Ag) & 15) == 0);
  const bool aQM = (rsA == 1) && ((csA & 3) == 0) && ((((unsigned long)Ag) & 15) == 0);
  const bool bQK = (rsB == 1) && ((csB & 3) == 0) && (bShift == 0) &&
                   ((((unsigned long)Bg) & 15) == 0);
  const bool bQN = (csB == 1) && ((rsB & 3) == 0) && (((n0 + bShift) & 3) == 0) &&
                   ((((unsigned long)Bg) & 15) == 0);

  v8f acc[4];
#pragma unroll
  for (int t = 0; t < 4; ++t)
#pragma unroll
    for (int r = 0; r < 8; ++r) acc[t][r] = 0.f;

  for (int k0 = 0; k0 < K; k0 += TK) {
    const bool kin = (k0 + TK <= K);

    // ---------------- stage A ----------------
    if (kin && aFull && aQK) {              // k-contiguous: b128 load, b64 LDS store
#pragma unroll
      for (int j = 0; j < 4; ++j) {
        int i = tid + j * GT;               // 0..511 quads
        int m = i >> 3, kq = (i & 7) << 2;
        v4f v = *(const v4f*)(Ag + (long)(m0 + m) * rsA + (k0 + kq));
        v4u p; p[0] = f2bf(v[0]); p[1] = f2bf(v[1]); p[2] = f2bf(v[2]); p[3] = f2bf(v[3]);
        *(v4u*)(&Al[m][kq]) = p;
      }
    } else if (kin && aFull && aQM) {       // m-contiguous (Gram): b128 load over m
#pragma unroll
      for (int j = 0; j < 4; ++j) {
        int i = tid + j * GT;
        int k = i >> 4, mq = (i & 15) << 2;
        v4f v = *(const v4f*)(Ag + (long)(k0 + k) * csA + (m0 + mq));
        Al[mq + 0][k] = f2bf(v[0]); Al[mq + 1][k] = f2bf(v[1]);
        Al[mq + 2][k] = f2bf(v[2]); Al[mq + 3][k] = f2bf(v[3]);
      }
    } else if (kin && aFull) {              // full tile, strided: unguarded scalar
#pragma unroll
      for (int j = 0; j < 16; ++j) {
        int i = tid + j * GT;
        int m = i >> 5, k = i & 31;
        Al[m][k] = f2bf(Ag[(long)(m0 + m) * rsA + (long)(k0 + k) * csA]);
      }
    } else {                                // edge tile: guarded, zero-fill
      for (int i = tid; i < TM * TK; i += GT) {
        int m = i >> 5, k = i & 31;
        int gm = m0 + m, gk = k0 + k;
        float v = 0.f;
        if (gm < M && gk < K) v = Ag[(long)gm * rsA + (long)gk * csA];
        Al[m][k] = f2bf(v);
      }
    }

    // ---------------- stage B (transposed) ----------------
    if (kin && bFull && bQK) {              // k-contiguous: b128 load, b64 LDS store
#pragma unroll
      for (int j = 0; j < 4; ++j) {
        int i = tid + j * GT;
        int n = i >> 3, kq = (i & 7) << 2;
        v4f v = *(const v4f*)(Bg + (long)(n0 + n) * csB + (k0 + kq));
        v4u p; p[0] = f2bf(v[0]); p[1] = f2bf(v[1]); p[2] = f2bf(v[2]); p[3] = f2bf(v[3]);
        *(v4u*)(&Bt[n][kq]) = p;
      }
    } else if (kin && bFull && bQN) {       // n-contiguous: b128 load over n
#pragma unroll
      for (int j = 0; j < 4; ++j) {
        int i = tid + j * GT;
        int k = i >> 4, nq = (i & 15) << 2;
        v4f v = *(const v4f*)(Bg + (long)(k0 + k) * rsB + (n0 + nq + bShift));
        Bt[nq + 0][k] = f2bf(v[0]); Bt[nq + 1][k] = f2bf(v[1]);
        Bt[nq + 2][k] = f2bf(v[2]); Bt[nq + 3][k] = f2bf(v[3]);
      }
    } else if (kin && bFull) {              // full tile (e.g. shifted conv taps)
#pragma unroll
      for (int j = 0; j < 16; ++j) {
        int i = tid + j * GT;
        int k = i >> 6, n = i & 63;
        Bt[n][k] = f2bf(Bg[(long)(k0 + k) * rsB + (long)(n0 + n + bShift) * csB]);
      }
    } else {                                // edge tile: guarded, zero-fill
      for (int i = tid; i < TK * TN; i += GT) {
        int k = i >> 6, n = i & 63;
        int gk = k0 + k, gn = n0 + n + bShift;
        float v = 0.f;
        if (gk < K && gn >= 0 && gn < N) v = Bg[(long)gk * rsB + (long)gn * csB];
        Bt[n][k] = f2bf(v);
      }
    }
    __syncthreads();

    // A fragment (ISA 7.12.2, 16-bit A 16x32): lane(m=l16, half),
    // element e -> k = (e/8)*16 + half*8 + (e%8): two contiguous 8-elt chunks.
    FragU af;
    {
      const unsigned short* ar = &Al[wave * 16 + l16][0];
      v8u a0 = *(const v8u*)(ar + half * 8);        // k = half*8 ..
      v8u a1 = *(const v8u*)(ar + 16 + half * 8);   // k = 16+half*8 ..
      af.u = __builtin_shufflevector(a0, a1, 0, 1, 2, 3, 4, 5, 6, 7,
                                     8, 9, 10, 11, 12, 13, 14, 15);
    }

#pragma unroll
    for (int t = 0; t < 4; ++t) {
      // B fragment (32x16): lane(n=l16, half), element e -> k = (e/8)*16 + half*8 + (e%8)
      FragU bf;
      const unsigned short* br = &Bt[t * 16 + l16][0];
      v8u b0 = *(const v8u*)(br + half * 8);
      v8u b1 = *(const v8u*)(br + 16 + half * 8);
      bf.u = __builtin_shufflevector(b0, b1, 0, 1, 2, 3, 4, 5, 6, 7,
                                     8, 9, 10, 11, 12, 13, 14, 15);
      acc[t] = __builtin_amdgcn_wmma_f32_16x16x32_bf16(
          false, af.b, false, bf.b, (short)0, acc[t], false, false);
    }
    __syncthreads();
  }

  // C/D layout: vgpr r -> m = half*8 + r, n = l16
#pragma unroll
  for (int t = 0; t < 4; ++t) {
    int gn = n0 + t * 16 + l16;
    if (gn >= N) continue;
#pragma unroll
    for (int r = 0; r < 8; ++r) {
      int gm = m0 + wave * 16 + half * 8 + r;
      if (gm >= M) continue;
      float v = alpha * acc[t][r];
      if (biasMode == 1) v += bias[gn];
      else if (biasMode == 2) v += bias[gm];
      long ci = (long)gm * ldc + gn;
      if (accum) v += Cg[ci];
      Cg[ci] = v;
    }
  }
}

// ---------------------------------------------------------------------------
// Elementwise / reduction kernels
// ---------------------------------------------------------------------------

// BatchNorm over (B,L) per channel c, optional relu(1)/tanh(2). x:[B,C,L]
__global__ void bn_kernel(const float* __restrict__ x, float* __restrict__ y,
                          const float* __restrict__ g, const float* __restrict__ b,
                          int Bn, int C, int L, int act)
{
  __shared__ float s1[256], s2[256];
  const int c = blockIdx.x;
  const int tid = threadIdx.x;
  const long n = (long)Bn * L;
  float sum = 0.f, sq = 0.f;
  for (long i = tid; i < n; i += 256) {
    long bb = i / L, l = i % L;
    float v = x[(bb * C + c) * (long)L + l];
    sum += v; sq += v * v;
  }
  s1[tid] = sum; s2[tid] = sq; __syncthreads();
  for (int s = 128; s > 0; s >>= 1) {
    if (tid < s) { s1[tid] += s1[tid + s]; s2[tid] += s2[tid + s]; }
    __syncthreads();
  }
  const float mean = s1[0] / (float)n;
  const float var  = s2[0] / (float)n - mean * mean;
  const float rstd = rsqrtf(var + 1e-5f);
  const float gg = g[c], bb2 = b[c];
  for (long i = tid; i < n; i += 256) {
    long bi = i / L, l = i % L;
    long idx = (bi * C + c) * (long)L + l;
    float v = gg * (x[idx] - mean) * rstd + bb2;
    if (act == 1)      v = fmaxf(v, 0.f);
    else if (act == 2) v = tanhf(v);
    y[idx] = v;
  }
}

// rowwise softmax (in place); one block per row
__global__ void softmax_rows(float* __restrict__ x, int cols)
{
  __shared__ float red[128];
  float* p = x + (long)blockIdx.x * cols;
  const int tid = threadIdx.x;
  float mx = -3.4e38f;
  for (int i = tid; i < cols; i += 128) mx = fmaxf(mx, p[i]);
  red[tid] = mx; __syncthreads();
  for (int s = 64; s > 0; s >>= 1) {
    if (tid < s) red[tid] = fmaxf(red[tid], red[tid + s]);
    __syncthreads();
  }
  mx = red[0]; __syncthreads();
  float sum = 0.f;
  for (int i = tid; i < cols; i += 128) { float e = expf(p[i] - mx); p[i] = e; sum += e; }
  red[tid] = sum; __syncthreads();
  for (int s = 64; s > 0; s >>= 1) {
    if (tid < s) red[tid] += red[tid + s];
    __syncthreads();
  }
  const float inv = 1.f / red[0];
  for (int i = tid; i < cols; i += 128) p[i] *= inv;
}

__global__ void maxpool2_k(const float* __restrict__ x, float* __restrict__ y,
                           int L, int total)  // total = B*C*(L/2)
{
  int i = blockIdx.x * blockDim.x + threadIdx.x;
  if (i >= total) return;
  const int Lh = L >> 1;
  int l = i % Lh; long bc = i / Lh;
  const float* p = x + bc * L + 2 * l;
  y[bc * Lh + l] = fmaxf(p[0], p[1]);
}

// sq[b*L+n] = sum_c x[b,c,n]^2
__global__ void rowsq_k(const float* __restrict__ x, float* __restrict__ sq,
                        int C, int L, int total)  // total = B*L
{
  int i = blockIdx.x * blockDim.x + threadIdx.x;
  if (i >= total) return;
  int b = i / L, n = i % L;
  const float* p = x + (long)b * C * L + n;
  float s = 0.f;
  for (int c = 0; c < C; ++c) { float v = p[(long)c * L]; s += v * v; }
  sq[i] = s;
}

// per-batch: 10 smallest d[n,m] = sq[n]+sq[m]+gram[n,m]  (gram = -2 f.f^T)
__global__ void topk_k(const float* __restrict__ gram, const float* __restrict__ sq,
                       int* __restrict__ idx, int L)
{
  int n = blockIdx.x * blockDim.x + threadIdx.x;
  if (n >= L) return;
  float best[KNN]; int bidx[KNN];
#pragma unroll
  for (int j = 0; j < KNN; ++j) { best[j] = 3.4e38f; bidx[j] = 0; }
  const float sn = sq[n];
  const float* gr = gram + (long)n * L;
  for (int m = 0; m < L; ++m) {
    float d = sn + sq[m] + gr[m];
    if (d < best[KNN - 1]) {
      int j = KNN - 1;
      while (j > 0 && d < best[j - 1]) { best[j] = best[j - 1]; bidx[j] = bidx[j - 1]; --j; }
      best[j] = d; bidx[j] = m;
    }
  }
#pragma unroll
  for (int j = 0; j < KNN; ++j) idx[(long)n * KNN + j] = bidx[j];
}

// out[b,c,l] = mean_j x[b,c,idx[b,l,j]]
__global__ void gather_mean_k(const float* __restrict__ x, const int* __restrict__ idx,
                              float* __restrict__ out, int C, int L, int total)
{
  int i = blockIdx.x * blockDim.x + threadIdx.x;
  if (i >= total) return;
  int l = i % L; int t = i / L; int c = t % C; int b = t / C;
  const int* ip = idx + ((long)b * L + l) * KNN;
  const float* xp = x + ((long)b * C + c) * (long)L;
  float s = 0.f;
#pragma unroll
  for (int j = 0; j < KNN; ++j) s += xp[ip[j]];
  out[i] = s * (1.f / KNN);
}

// out[b,c,2l]=x[b,c,l]; out[b,c,2l+1]=kn[b,c,l]  (out batch stride = obs)
__global__ void interleave_k(const float* __restrict__ x, const float* __restrict__ kn,
                             float* __restrict__ out, int C, int L, long obs, int total)
{
  int i = blockIdx.x * blockDim.x + threadIdx.x;
  if (i >= total) return;
  int l = i % L; int t = i / L; int c = t % C; int b = t / C;
  float* o = out + (long)b * obs + (long)c * (2 * L);
  o[2 * l]     = x[i];
  o[2 * l + 1] = kn[i];
}

// ---------------------------------------------------------------------------
// Host-side helpers
// ---------------------------------------------------------------------------
struct Ws {
  float *ping, *pong, *mid;
  float *qh, *kh, *vh, *sc, *ob;   // kh doubles as Gram, vh doubles as knn-out
  float *x1u, *x2u, *x2b, *x3;
  float *sq;
  int   *idx;
};

static void gemm(hipStream_t st, const float* A, const float* B, const float* bias,
                 float* C, int M, int N, int K,
                 int rsA, int csA, int rsB, int csB, int ldc,
                 long aOut, long aIn, long bOut, long bIn, long cOut, long cIn,
                 int outer, int inner, float alpha, int bShift, int biasMode, int accum)
{
  dim3 g((N + TN - 1) / TN, (M + TM - 1) / TM, outer * inner);
  gemm_wmma<<<g, GT, 0, st>>>(A, B, bias, C, M, N, K, rsA, csA, rsB, csB, ldc,
                              aOut, aIn, bOut, bIn, cOut, cIn,
                              inner, alpha, bShift, biasMode, accum);
}

// conv1d (NCH, OIH), pad=(ks-1)/2, as ks shifted accumulating GEMMs per batch
static void conv_run(hipStream_t st, const float* x, int cin, int L,
                     const float* w, const float* b, int cout, int ks, float* y)
{
  int pad = (ks - 1) / 2;
  for (int d = 0; d < ks; ++d) {
    gemm(st, w + d, x, b, y, cout, L, cin,
         /*rsA*/ cin * ks, /*csA*/ ks, /*rsB*/ L, /*csB*/ 1, /*ldc*/ L,
         0, 0, (long)cin * L, 0, (long)cout * L, 0,
         BAT, 1, 1.f, d - pad, (d == 0) ? 2 : 0, (d == 0) ? 0 : 1);
  }
}

static void bn_run(hipStream_t st, const float* x, float* y, const float* g,
                   const float* b, int C, int L, int act)
{
  bn_kernel<<<C, 256, 0, st>>>(x, y, g, b, BAT, C, L, act);
}

struct MhaP { const float *in_w, *in_b, *out_w, *out_b; };

static void mha_run(hipStream_t st, const Ws& ws,
                    const float* q, int q_rs, long q_bs, int Sq,
                    const float* kv, int kv_rs, long kv_bs, int Skv,
                    int E, MhaP p, float* dest, int d_ldc, long d_bs)
{
  const int hd = E / HEADS;
  // projections: y[s,o] = t[s,:].w[o,:] + b[o]   (w rows: q [0,E), k [E,2E), v [2E,3E))
  gemm(st, q,  p.in_w + 0L * E * E, p.in_b + 0L * E, ws.qh, Sq,  E, E,
       q_rs, 1, 1, E, E,  q_bs, 0, 0, 0, (long)Sq * E, 0, BAT, 1, 1.f, 0, 1, 0);
  gemm(st, kv, p.in_w + 1L * E * E, p.in_b + 1L * E, ws.kh, Skv, E, E,
       kv_rs, 1, 1, E, E, kv_bs, 0, 0, 0, (long)Skv * E, 0, BAT, 1, 1.f, 0, 1, 0);
  gemm(st, kv, p.in_w + 2L * E * E, p.in_b + 2L * E, ws.vh, Skv, E, E,
       kv_rs, 1, 1, E, E, kv_bs, 0, 0, 0, (long)Skv * E, 0, BAT, 1, 1.f, 0, 1, 0);
  // scores[b,h] = Q_h K_h^T * scale   (z over B*H)
  const float scale = 1.f / sqrtf((float)hd);
  gemm(st, ws.qh, ws.kh, nullptr, ws.sc, Sq, Skv, hd,
       E, 1, 1, E, Skv,
       (long)Sq * E, hd, (long)Skv * E, hd,
       (long)HEADS * Sq * Skv, (long)Sq * Skv,
       BAT, HEADS, scale, 0, 0, 0);
  softmax_rows<<<BAT * HEADS * Sq, 128, 0, st>>>(ws.sc, Skv);
  // o[b,h] = A V_h, written as [B,Sq,E] with head offset
  gemm(st, ws.sc, ws.vh, nullptr, ws.ob, Sq, hd, Skv,
       Skv, 1, E, 1, E,
       (long)HEADS * Sq * Skv, (long)Sq * Skv, (long)Skv * E, hd,
       (long)Sq * E, hd,
       BAT, HEADS, 1.f, 0, 0, 0);
  // out proj
  gemm(st, ws.ob, p.out_w, p.out_b, dest, Sq, E, E,
       E, 1, 1, E, d_ldc,
       (long)Sq * E, 0, 0, 0, d_bs, 0, BAT, 1, 1.f, 0, 1, 0);
}

// kNN mean: Gram via WMMA GEMM (alpha=-2, strided X^T X), then top-10, gather-mean
static void knn_run(hipStream_t st, const Ws& ws, const float* x, int C, int L, float* out)
{
  int totBL = BAT * L;
  rowsq_k<<<(totBL + 255) / 256, 256, 0, st>>>(x, ws.sq, C, L, totBL);
  for (int b = 0; b < BAT; ++b) {
    const float* xb = x + (long)b * C * L;
    gemm(st, xb, xb, nullptr, ws.kh, L, L, C,
         1, L, L, 1, L, 0, 0, 0, 0, 0, 0, 1, 1, -2.f, 0, 0, 0);
    topk_k<<<(L + 127) / 128, 128, 0, st>>>(ws.kh, ws.sq + (long)b * L,
                                            ws.idx + (long)b * L * KNN, L);
  }
  int tot = BAT * C * L;
  gather_mean_k<<<(tot + 255) / 256, 256, 0, st>>>(x, ws.idx, out, C, L, tot);
}

// ---------------------------------------------------------------------------
extern "C" void kernel_launch(void* const* d_in, const int* in_sizes, int n_in,
                              void* d_out, int out_size, void* d_ws, size_t ws_size,
                              hipStream_t stream)
{
  (void)in_sizes; (void)n_in; (void)out_size; (void)ws_size;
  auto in = [&](int i) { return (const float*)d_in[i]; };

  // ---- workspace (bump allocator; ~167MB fp32) ----
  float* base = (float*)d_ws;
  size_t o = 0;
  auto alloc = [&](size_t n) { float* p = base + o; o += n; return p; };
  Ws ws;
  ws.ping = alloc(8388608);  ws.pong = alloc(8388608);  ws.mid = alloc(4194304);
  ws.qh = alloc(2097152);    ws.kh = alloc(4194304);    ws.vh = alloc(4194304);
  ws.sc = alloc(1048576);    ws.ob = alloc(2097152);
  ws.x1u = alloc(262144);    ws.x2u = alloc(524288);
  ws.x2b = alloc(2097152);   ws.x3  = alloc(4194304);
  ws.sq  = alloc(8192);
  ws.idx = (int*)(base + o); o += BAT * 2048 * KNN;

  hipStream_t st = stream;

  // ---- attn0: mha(x, x)  (B,3,1024), S=3, E=1024 -> ping ----
  mha_run(st, ws, in(0), 1024, 3072, 3, in(0), 1024, 3072, 3, 1024,
          {in(1), in(2), in(3), in(4)}, ws.ping, 1024, 3072);

  // ---- enc1: 3->32 (ks3) -> 32->64 (ks1); x1u saved; pool -> ping ----
  conv_run(st, ws.ping, 3, 1024, in(5), in(6), 32, 3, ws.mid);
  bn_run(st, ws.mid, ws.mid, in(7), in(8), 32, 1024, 1);
  conv_run(st, ws.mid, 32, 1024, in(9), in(10), 64, 1, ws.x1u);
  bn_run(st, ws.x1u, ws.x1u, in(11), in(12), 64, 1024, 1);
  maxpool2_k<<<(BAT * 64 * 512 + 255) / 256, 256, 0, st>>>(ws.x1u, ws.ping, 1024, BAT * 64 * 512);

  // ---- enc2: 64->128 (ks3) -> 128->256 (ks1); x2u saved; pool -> ping ----
  conv_run(st, ws.ping, 64, 512, in(13), in(14), 128, 3, ws.mid);
  bn_run(st, ws.mid, ws.mid, in(15), in(16), 128, 512, 1);
  conv_run(st, ws.mid, 128, 512, in(17), in(18), 256, 1, ws.x2u);
  bn_run(st, ws.x2u, ws.x2u, in(19), in(20), 256, 512, 1);
  maxpool2_k<<<(BAT * 256 * 256 + 255) / 256, 256, 0, st>>>(ws.x2u, ws.ping, 512, BAT * 256 * 256);

  // ---- kattdec(x2, up1_attn, x2u): cat(y,e) built directly in pong (4,512,512) ----
  knn_run(st, ws, ws.ping, 256, 256, ws.vh);
  interleave_k<<<(BAT * 256 * 256 + 255) / 256, 256, 0, st>>>(ws.ping, ws.vh, ws.pong,
                                                              256, 256, 512L * 512, BAT * 256 * 256);
  mha_run(st, ws, ws.pong, 512, 512L * 512, 256,          // q = y (S=256,E=512)
          ws.x2u, 512, 256L * 512, 256, 512,              // kv = x2u
          {in(21), in(22), in(23), in(24)},
          ws.pong + 256L * 512, 512, 512L * 512);         // e -> channels [256,512)

  // ---- down1: 512->512 (ks3) -> 512->512 (ks1) -> ping (4,512,512) ----
  conv_run(st, ws.pong, 512, 512, in(25), in(26), 512, 3, ws.mid);
  bn_run(st, ws.mid, ws.mid, in(27), in(28), 512, 512, 1);
  conv_run(st, ws.mid, 512, 512, in(29), in(30), 512, 1, ws.ping);
  bn_run(st, ws.ping, ws.ping, in(31), in(32), 512, 512, 1);

  // ---- kattdec(x, up2_attn, x1u): cat in pong (4,1024,1024) ----
  knn_run(st, ws, ws.ping, 512, 512, ws.vh);
  interleave_k<<<(BAT * 512 * 512 + 255) / 256, 256, 0, st>>>(ws.ping, ws.vh, ws.pong,
                                                              512, 512, 1024L * 1024, BAT * 512 * 512);
  mha_run(st, ws, ws.pong, 1024, 1024L * 1024, 512,       // q = y (S=512,E=1024)
          ws.x1u, 1024, 64L * 1024, 64, 1024,             // kv = x1u (S=64)
          {in(33), in(34), in(35), in(36)},
          ws.pong + 512L * 1024, 1024, 1024L * 1024);

  // ---- down2: 1024->512 (ks3) -> 512->512 (ks1) -> x2b (4,512,1024) ----
  conv_run(st, ws.pong, 1024, 1024, in(37), in(38), 512, 3, ws.mid);
  bn_run(st, ws.mid, ws.mid, in(39), in(40), 512, 1024, 1);
  conv_run(st, ws.mid, 512, 1024, in(41), in(42), 512, 1, ws.x2b);
  bn_run(st, ws.x2b, ws.x2b, in(43), in(44), 512, 1024, 1);

  // ---- kattdec(x2b, None) -> ping (4,512,2048); down3 -> x3 ----
  knn_run(st, ws, ws.x2b, 512, 1024, ws.vh);
  interleave_k<<<(BAT * 512 * 1024 + 255) / 256, 256, 0, st>>>(ws.x2b, ws.vh, ws.ping,
                                                               512, 1024, 512L * 2048, BAT * 512 * 1024);
  conv_run(st, ws.ping, 512, 2048, in(45), in(46), 512, 3, ws.mid);
  bn_run(st, ws.mid, ws.mid, in(47), in(48), 512, 2048, 1);
  conv_run(st, ws.mid, 512, 2048, in(49), in(50), 512, 1, ws.x3);
  bn_run(st, ws.x3, ws.x3, in(51), in(52), 512, 2048, 1);

  // ---- kattdec(x3, None) -> ping (4,512,4096); down4 -> pong (4,256,4096) ----
  knn_run(st, ws, ws.x3, 512, 2048, ws.vh);
  interleave_k<<<(BAT * 512 * 2048 + 255) / 256, 256, 0, st>>>(ws.x3, ws.vh, ws.ping,
                                                               512, 2048, 512L * 4096, BAT * 512 * 2048);
  conv_run(st, ws.ping, 512, 4096, in(53), in(54), 256, 3, ws.mid);
  bn_run(st, ws.mid, ws.mid, in(55), in(56), 256, 4096, 1);
  conv_run(st, ws.mid, 256, 4096, in(57), in(58), 256, 1, ws.pong);
  bn_run(st, ws.pong, ws.pong, in(59), in(60), 256, 4096, 1);

  // ---- fc1 (pool): 256->128->128 (ks1); pool -> pong (4,128,2048) ----
  conv_run(st, ws.pong, 256, 4096, in(61), in(62), 128, 1, ws.mid);
  bn_run(st, ws.mid, ws.mid, in(63), in(64), 128, 4096, 1);
  conv_run(st, ws.mid, 128, 4096, in(65), in(66), 128, 1, ws.ping);
  bn_run(st, ws.ping, ws.ping, in(67), in(68), 128, 4096, 1);
  maxpool2_k<<<(BAT * 128 * 2048 + 255) / 256, 256, 0, st>>>(ws.ping, ws.pong, 4096, BAT * 128 * 2048);

  // ---- attn2: mha(x, x3) -> ping (4,128,2048) ----
  mha_run(st, ws, ws.pong, 2048, 128L * 2048, 128,
          ws.x3, 2048, 512L * 2048, 512, 2048,
          {in(69), in(70), in(71), in(72)}, ws.ping, 2048, 128L * 2048);

  // ---- fc2 (pool): 128->64->32 (ks1); pool -> ping (4,32,1024) ----
  conv_run(st, ws.ping, 128, 2048, in(73), in(74), 64, 1, ws.mid);
  bn_run(st, ws.mid, ws.mid, in(75), in(76), 64, 2048, 1);
  conv_run(st, ws.mid, 64, 2048, in(77), in(78), 32, 1, ws.pong);
  bn_run(st, ws.pong, ws.pong, in(79), in(80), 32, 2048, 1);
  maxpool2_k<<<(BAT * 32 * 1024 + 255) / 256, 256, 0, st>>>(ws.pong, ws.ping, 2048, BAT * 32 * 1024);

  // ---- attn3: mha(x, x2b) -> pong (4,32,1024) ----
  mha_run(st, ws, ws.ping, 1024, 32L * 1024, 32,
          ws.x2b, 1024, 512L * 1024, 512, 1024,
          {in(81), in(82), in(83), in(84)}, ws.pong, 1024, 32L * 1024);

  // ---- fc3 (tanh): 32->16->3 (ks1); final bn+tanh -> d_out (4,3,1024) ----
  conv_run(st, ws.pong, 32, 1024, in(85), in(86), 16, 1, ws.mid);
  bn_run(st, ws.mid, ws.mid, in(87), in(88), 16, 1024, 1);
  conv_run(st, ws.mid, 16, 1024, in(89), in(90), 3, 1, ws.ping);
  bn_run(st, ws.ping, (float*)d_out, in(91), in(92), 3, 1024, 2);
}
